// Combiner_75359496175816
// MI455X (gfx1250) — matrix operand order repaired and checked by reading, not compiled, not run on hardware
//
#include <hip/hip_runtime.h>
#include <hip/hip_bf16.h>

#define B_ 16
#define N_ 2048
#define C_ 768
#define CL 2

typedef __attribute__((ext_vector_type(16))) __bf16 v16bf;
typedef __attribute__((ext_vector_type(8)))  __bf16 v8bf;
typedef __attribute__((ext_vector_type(8)))  float  v8f;
typedef __attribute__((ext_vector_type(4)))  float  f4;

__device__ inline v16bf cat8(v8bf lo, v8bf hi) {
  return __builtin_shufflevector(lo, hi, 0, 1, 2, 3, 4, 5, 6, 7,
                                         8, 9, 10, 11, 12, 13, 14, 15);
}

// ---- direct-from-memory fragment loads (16x16x32 bf16, this file's contiguous k = 16h + s order) ---
// A (16x32 MxK), row-major source, row stride C_:
//   lane: row = lane&15, k-half = (lane>>4)<<4; 16 contiguous k. Two b128 loads.
__device__ inline v16bf frag_a_mem(const __bf16* base /* row0,k0 */, int lane) {
  const __bf16* p = base + (size_t)(lane & 15) * C_ + ((lane >> 4) << 4);
  const v8bf lo = *(const v8bf*)p;
  const v8bf hi = *(const v8bf*)(p + 8);
  return cat8(lo, hi);
}
// B (32x16 KxN) where B[k][n] = X[n][k] (row-major X, stride C_):
//   lane: n = lane&15, k-half = (lane>>4)<<4; 16 contiguous k. Two b128 loads.
__device__ inline v16bf frag_b_mem(const __bf16* base /* n0 row, k0 */, int lane) {
  const __bf16* p = base + (size_t)(lane & 15) * C_ + ((lane >> 4) << 4);
  const v8bf lo = *(const v8bf*)p;
  const v8bf hi = *(const v8bf*)(p + 8);
  return cat8(lo, hi);
}
// Fragment already materialized contiguously (32-byte aligned): two b128 loads.
__device__ inline v16bf frag_flat(const __bf16* p) {
  return cat8(*(const v8bf*)p, *(const v8bf*)(p + 8));
}

// ---------------- P0a: bf16 copy of x with one zero pad row per batch edge ----------
// xp layout: [B][N+2][C]; token t at row t+1; rows 0 and N+1 are zero.
__global__ __launch_bounds__(256) void k_xpad(const float* __restrict__ x,
                                              __bf16* __restrict__ xp) {
  const size_t i8 = ((size_t)blockIdx.x * 256 + threadIdx.x) * 8;
  if (i8 >= (size_t)B_ * (N_ + 2) * C_) return;
  const int col = (int)(i8 % C_);
  const size_t row = i8 / C_;
  const int b = (int)(row / (N_ + 2));
  const int r = (int)(row % (N_ + 2));
  v8bf o;
  if (r == 0 || r == N_ + 1) {
#pragma unroll
    for (int e = 0; e < 8; ++e) o[e] = (__bf16)0.f;
  } else {
    const float* p = x + ((size_t)b * N_ + (r - 1)) * C_ + col;
    const f4 q0 = *(const f4*)p;
    const f4 q1 = *(const f4*)(p + 4);
#pragma unroll
    for (int e = 0; e < 4; ++e) { o[e] = (__bf16)q0[e]; o[e + 4] = (__bf16)q1[e]; }
  }
  *(v8bf*)(xp + i8) = o;
}

// ---------------- P0b: pre-swizzle conv weights into fragment order (bf16) ----------
// wfrag[s][c0blk=24][kblk=24][nblk=2][lane=32][e=16]
__global__ __launch_bounds__(256) void k_wprep(const float* __restrict__ w,
                                               __bf16* __restrict__ wfrag) {
  const int gid = blockIdx.x * 256 + threadIdx.x;
  if (gid >= 3 * C_ * C_) return;
  const int s  = gid / (C_ * C_);
  const int r  = gid % (C_ * C_);
  const int co = r / C_;
  const int ci = r % C_;
  const int c0blk = co >> 5, n  = co & 31;
  const int kblk  = ci >> 5, kl = ci & 31;
  const int nblk = n >> 4;
  const int lane = (n & 15) + (((kl >> 4) & 1) << 4);
  const int e    = kl & 15;
  wfrag[(((((size_t)s * 24 + c0blk) * 24 + kblk) * 2 + nblk) * 32 + lane) * 16 + e] =
      (__bf16)w[((size_t)co * C_ + ci) * 3 + s];
}

// ---------------- K1: residual token conv — barrier-free, LDS-free WMMA GEMM --------
__global__ __launch_bounds__(256) void k_conv(const float* __restrict__ x,
                                              const __bf16* __restrict__ xp,
                                              const __bf16* __restrict__ wfrag,
                                              float* __restrict__ y) {
  const int c0 = blockIdx.x * 32;
  const int n0 = blockIdx.y * 64;
  const int b  = blockIdx.z;
  const int tid  = threadIdx.x;
  const int lane = tid & 31;
  const int wave = tid >> 5;
  const int wm = wave >> 1, wn = wave & 1;
  v8f acc = {};
  for (int s = 0; s < 3; ++s) {
    // A rows: padded layout makes the shift branch-free (token t at row t+1).
    const __bf16* ab = xp + ((size_t)b * (N_ + 2) + n0 + wm * 16 + s) * C_;
    const __bf16* wb = wfrag + ((((size_t)s * 24 + (c0 >> 5)) * 24) * 2 + wn) * 512
                             + (size_t)lane * 16;
#pragma unroll 4
    for (int kc = 0; kc < C_; kc += 32) {
      const v16bf a  = frag_a_mem(ab + kc, lane);
      const v16bf bv = frag_flat(wb + (size_t)(kc >> 5) * 1024);
      acc = __builtin_amdgcn_wmma_f32_16x16x32_bf16(false, a, false, bv,
                                                    (short)0, acc, false, false);
    }
  }
#pragma unroll
  for (int r = 0; r < 8; ++r) {
    const int m = wm * 16 + r + ((lane >> 4) << 3);
    const int n = wn * 16 + (lane & 15);
    const size_t o = ((size_t)b * N_ + (n0 + m)) * C_ + (c0 + n);
    y[o] = acc[r] + x[o];
  }
}

// ---------------- K2: fused LayerNorm + score weight + sumsq + bf16 mirror ----------
__global__ __launch_bounds__(256) void k_ln(float* __restrict__ y,
                                            const float* __restrict__ gamma,
                                            const float* __restrict__ beta,
                                            const float* __restrict__ sw,
                                            const float* __restrict__ sb,
                                            __bf16* __restrict__ xbf,
                                            float* __restrict__ tw,
                                            float* __restrict__ sq) {
  __shared__ float red[256];
  const int tid = threadIdx.x;
  const size_t row = blockIdx.x;                        // b*N + n
  float* p = y + row * C_;
  float v[3];
  float s1 = 0.f, s2 = 0.f;
#pragma unroll
  for (int i = 0; i < 3; ++i) { v[i] = p[tid + i * 256]; s1 += v[i]; s2 += v[i] * v[i]; }
  red[tid] = s1; __syncthreads();
  for (int st = 128; st > 0; st >>= 1) { if (tid < st) red[tid] += red[tid + st]; __syncthreads(); }
  const float mu = red[0] * (1.f / C_); __syncthreads();
  red[tid] = s2; __syncthreads();
  for (int st = 128; st > 0; st >>= 1) { if (tid < st) red[tid] += red[tid + st]; __syncthreads(); }
  const float var = red[0] * (1.f / C_) - mu * mu; __syncthreads();
  const float rstd = rsqrtf(var + 1e-5f);
  float sc = 0.f, qs = 0.f;
#pragma unroll
  for (int i = 0; i < 3; ++i) {
    const int c = tid + i * 256;
    const float xn = (v[i] - mu) * rstd * gamma[c] + beta[c];
    p[c] = xn;
    xbf[row * C_ + c] = (__bf16)xn;
    sc += xn * sw[c];
    qs += xn * xn;
  }
  red[tid] = sc; __syncthreads();
  for (int st = 128; st > 0; st >>= 1) { if (tid < st) red[tid] += red[tid + st]; __syncthreads(); }
  const float score = red[0] + sb[0]; __syncthreads();
  red[tid] = qs; __syncthreads();
  for (int st = 128; st > 0; st >>= 1) { if (tid < st) red[tid] += red[tid + st]; __syncthreads(); }
  if (tid == 0) { tw[row] = expf(score); sq[row] = red[0]; }
}

// ------- 64x32 distance tile: barrier-free WMMA K-loop, fragments direct from HBM ---
__device__ inline void dist_tile(const __bf16* __restrict__ xb, int i0, int j0,
                                 const float* sqi, const float* sqj,
                                 float (*Dt)[33], int tid) {
  const int lane = tid & 31;
  const int wave = tid >> 5;
  const int wm = wave >> 1, wn = wave & 1;
  const __bf16* ab = xb + (size_t)(i0 + wm * 16) * C_;
  const __bf16* bb = xb + (size_t)(j0 + wn * 16) * C_;
  v8f acc = {};
#pragma unroll 4
  for (int kc = 0; kc < C_; kc += 32) {
    const v16bf a  = frag_a_mem(ab + kc, lane);
    const v16bf bv = frag_b_mem(bb + kc, lane);
    acc = __builtin_amdgcn_wmma_f32_16x16x32_bf16(false, a, false, bv,
                                                  (short)0, acc, false, false);
  }
  const float rc = rsqrtf((float)C_);
#pragma unroll
  for (int r = 0; r < 8; ++r) {
    const int m = wm * 16 + r + ((lane >> 4) << 3);
    const int n = wn * 16 + (lane & 15);
    const float d2 = sqi[m] + sqj[n] - 2.f * acc[r];
    Dt[m][n] = sqrtf(fmaxf(d2, 0.f)) * rc;
  }
  __syncthreads();
}

// ---------------- K3a: 3-NN density sweep (recomputes Gram tiles) -------------------
__global__ __launch_bounds__(256) void k_density(const __bf16* __restrict__ xbf,
                                                 const float* __restrict__ sq,
                                                 const float* __restrict__ noise,
                                                 float* __restrict__ density,
                                                 float* __restrict__ rowmax) {
  __shared__ float Dt[64][33];
  __shared__ float sqi[64];
  __shared__ float sqj[32];
  __shared__ float rmaxs[64];
  const int b  = blockIdx.y;
  const int i0 = blockIdx.x * 64;
  const int tid = threadIdx.x;
  const __bf16* xb = xbf + (size_t)b * N_ * C_;
  if (tid < 64) sqi[tid] = sq[b * N_ + i0 + tid];
  float t0 = 3.4e38f, t1 = 3.4e38f, t2 = 3.4e38f, rmax = 0.f;
  for (int j0 = 0; j0 < N_; j0 += 32) {
    if (tid < 32) sqj[tid] = sq[b * N_ + j0 + tid];
    __syncthreads();
    dist_tile(xb, i0, j0, sqi, sqj, Dt, tid);
    if (tid < 64) {
#pragma unroll
      for (int c = 0; c < 32; ++c) {
        const float d = Dt[tid][c];
        rmax = fmaxf(rmax, d);
        if (d < t0)      { t2 = t1; t1 = t0; t0 = d; }
        else if (d < t1) { t2 = t1; t1 = d; }
        else if (d < t2) { t2 = d; }
      }
    }
    __syncthreads();
  }
  if (tid < 64) {
    const int i = b * N_ + i0 + tid;
    density[i] = expf(-(t0 * t0 + t1 * t1 + t2 * t2) * (1.f / 3.f)) + noise[i] * 1e-6f;
    rmaxs[tid] = rmax;
  }
  __syncthreads();
  if (tid == 0) {
    float m = 0.f;
    for (int i = 0; i < 64; ++i) m = fmaxf(m, rmaxs[i]);
    rowmax[b * 32 + blockIdx.x] = m;
  }
}

// ---------------- K3b: density-masked min-distance sweep ----------------------------
__global__ __launch_bounds__(256) void k_distmin(const __bf16* __restrict__ xbf,
                                                 const float* __restrict__ sq,
                                                 const float* __restrict__ density,
                                                 float* __restrict__ dminm) {
  __shared__ float Dt[64][33];
  __shared__ float sqi[64];
  __shared__ float sqj[32];
  __shared__ float dnj[32];
  const int b  = blockIdx.y;
  const int i0 = blockIdx.x * 64;
  const int tid = threadIdx.x;
  const __bf16* xb = xbf + (size_t)b * N_ * C_;
  float di = 0.f;
  if (tid < 64) { sqi[tid] = sq[b * N_ + i0 + tid]; di = density[b * N_ + i0 + tid]; }
  float dmin = 3.4e38f;
  for (int j0 = 0; j0 < N_; j0 += 32) {
    if (tid < 32) { sqj[tid] = sq[b * N_ + j0 + tid]; dnj[tid] = density[b * N_ + j0 + tid]; }
    __syncthreads();
    dist_tile(xb, i0, j0, sqi, sqj, Dt, tid);
    if (tid < 64) {
#pragma unroll
      for (int c = 0; c < 32; ++c)
        if (dnj[c] > di) dmin = fminf(dmin, Dt[tid][c]);
    }
    __syncthreads();
  }
  if (tid < 64) dminm[b * N_ + i0 + tid] = dmin;
}

// ---------------- K4: per-batch dist_max reduce + top-2 of dist_min*density ---------
__global__ __launch_bounds__(256) void k_top2(const float* __restrict__ rowmax,
                                              const float* __restrict__ dminm,
                                              const float* __restrict__ density,
                                              int* __restrict__ index_down) {
  __shared__ float bv[256][2];
  __shared__ int   bi[256][2];
  __shared__ float dmax_s;
  const int b = blockIdx.x;
  const int tid = threadIdx.x;
  if (tid == 0) {
    float m = 0.f;
    for (int i = 0; i < 32; ++i) m = fmaxf(m, rowmax[b * 32 + i]);
    dmax_s = m;
  }
  __syncthreads();
  const float dmax = dmax_s;
  float v0 = -3.4e38f, v1 = -3.4e38f; int ix0 = -1, ix1 = -1;
  for (int n = tid; n < N_; n += 256) {
    const float s = fminf(dminm[b * N_ + n], dmax) * density[b * N_ + n];
    if (s > v0 || (s == v0 && n < ix0)) { v1 = v0; ix1 = ix0; v0 = s; ix0 = n; }
    else if (s > v1 || (s == v1 && n < ix1)) { v1 = s; ix1 = n; }
  }
  bv[tid][0] = v0; bv[tid][1] = v1; bi[tid][0] = ix0; bi[tid][1] = ix1;
  __syncthreads();
  if (tid == 0) {
    float w0 = -3.4e38f, w1 = -3.4e38f; int j0 = -1, j1 = -1;
    for (int t = 0; t < 256; ++t)
      for (int k = 0; k < 2; ++k) {
        const float s = bv[t][k]; const int id = bi[t][k];
        if (id < 0) continue;
        if (s > w0 || (s == w0 && id < j0)) { w1 = w0; j1 = j0; w0 = s; j0 = id; }
        else if (s > w1 || (s == w1 && id < j1)) { w1 = s; j1 = id; }
      }
    index_down[b * 2 + 0] = j0;
    index_down[b * 2 + 1] = j1;
  }
}

// ---------------- K5a: cluster assignment (wave per token) + weight seg-sum ---------
__global__ __launch_bounds__(256) void k_assign(const float* __restrict__ xln,
                                                const float* __restrict__ sq,
                                                const int* __restrict__ index_down,
                                                const float* __restrict__ tw,
                                                int* __restrict__ idxc,
                                                float* __restrict__ allw) {
  const int tid  = threadIdx.x;
  const int lane = tid & 31;
  const int wave = tid >> 5;
  const int tok = blockIdx.x * 8 + wave;                // global token over B*N
  const int b = tok / N_;
  const int n = tok % N_;
  const int c0 = index_down[b * 2 + 0];
  const int c1 = index_down[b * 2 + 1];
  const float* xi  = xln + (size_t)(b * N_ + n)  * C_;
  const float* xc0 = xln + (size_t)(b * N_ + c0) * C_;
  const float* xc1 = xln + (size_t)(b * N_ + c1) * C_;
  float dot0 = 0.f, dot1 = 0.f;
  for (int c = lane; c < C_; c += 32) {
    const float v = xi[c];
    dot0 += v * xc0[c];
    dot1 += v * xc1[c];
  }
#pragma unroll
  for (int o = 16; o > 0; o >>= 1) {
    dot0 += __shfl_down(dot0, o, 32);
    dot1 += __shfl_down(dot1, o, 32);
  }
  if (lane == 0) {
    const float d0 = sq[b * N_ + n] + sq[b * N_ + c0] - 2.f * dot0;
    const float d1 = sq[b * N_ + n] + sq[b * N_ + c1] - 2.f * dot1;
    int idx = (d1 < d0) ? 1 : 0;
    if (n == c0) idx = 0;
    if (n == c1) idx = 1;
    idxc[b * N_ + n] = idx;
    atomicAdd(&allw[b * 2 + idx], tw[b * N_ + n]);
  }
}

// ---------------- K5b: weighted merge (LDS-staged segment sum) ----------------------
__global__ __launch_bounds__(256) void k_merge(const float* __restrict__ xln,
                                               const float* __restrict__ tw,
                                               const float* __restrict__ allw,
                                               const int* __restrict__ idxc,
                                               float* __restrict__ out_x,
                                               float* __restrict__ out_agg,
                                               float* __restrict__ out_idx) {
  __shared__ float accs[2][C_];
  __shared__ float norm_s[256];
  __shared__ int   idx_s[256];
  const int tid = threadIdx.x;
  const int t0  = blockIdx.x * 256;                     // token base (all same batch)
  const int b   = t0 / N_;
  {
    const int tok = t0 + tid;
    const int ic  = idxc[tok];
    const float nw = tw[tok] / (allw[b * 2 + ic] + 1e-6f);
    norm_s[tid] = nw; idx_s[tid] = ic;
    out_agg[tok] = nw;
    out_idx[tok] = (float)ic;
  }
  for (int c = tid; c < 2 * C_; c += 256) (&accs[0][0])[c] = 0.f;
  __syncthreads();
  for (int t = 0; t < 256; ++t) {                       // channels owned per-thread
    const float nw = norm_s[t];
    const int   ic = idx_s[t];
    const float* xr = xln + (size_t)(t0 + t) * C_;
#pragma unroll
    for (int i = 0; i < 3; ++i) {
      const int c = tid + i * 256;
      accs[ic][c] += xr[c] * nw;
    }
  }
  __syncthreads();
  for (int c = tid; c < 2 * C_; c += 256)
    atomicAdd(&out_x[b * 2 * C_ + c], (&accs[0][0])[c]);
}

// ---------------- K0: zero the accumulated outputs ----------------------------------
__global__ void k_init(float* __restrict__ out_x, float* __restrict__ allw) {
  const int i = blockIdx.x * 256 + threadIdx.x;
  if (i < B_ * CL * C_) out_x[i] = 0.f;
  if (i < B_ * CL) allw[i] = 0.f;
}

extern "C" void kernel_launch(void* const* d_in, const int* in_sizes, int n_in,
                              void* d_out, int out_size, void* d_ws, size_t ws_size,
                              hipStream_t stream) {
  const float* x  = (const float*)d_in[0];
  const float* cw = (const float*)d_in[1];
  const float* g  = (const float*)d_in[2];
  const float* be = (const float*)d_in[3];
  const float* sw = (const float*)d_in[4];
  const float* sb = (const float*)d_in[5];
  const float* nz = (const float*)d_in[6];

  char* ws = (char*)d_ws;
  size_t off = 0;
  float*  y     = (float*)(ws + off);  off += (size_t)B_ * N_ * C_ * sizeof(float);
  __bf16* xbf   = (__bf16*)(ws + off); off += (size_t)B_ * N_ * C_ * sizeof(__bf16);
  __bf16* xpad  = (__bf16*)(ws + off); off += (size_t)B_ * (N_ + 2) * C_ * sizeof(__bf16);
  __bf16* wfrag = (__bf16*)(ws + off); off += (size_t)3 * C_ * C_ * sizeof(__bf16);
  float*  tw    = (float*)(ws + off);  off += (size_t)B_ * N_ * sizeof(float);
  float*  sq    = (float*)(ws + off);  off += (size_t)B_ * N_ * sizeof(float);
  float*  dens  = (float*)(ws + off);  off += (size_t)B_ * N_ * sizeof(float);
  float*  dmin  = (float*)(ws + off);  off += (size_t)B_ * N_ * sizeof(float);
  float*  rmax  = (float*)(ws + off);  off += (size_t)B_ * 32 * sizeof(float);
  int*    idn   = (int*)(ws + off);    off += (size_t)B_ * CL * sizeof(int);
  float*  alw   = (float*)(ws + off);  off += (size_t)B_ * CL * sizeof(float);
  int*    idc   = (int*)(ws + off);    off += (size_t)B_ * N_ * sizeof(int);

  float* out     = (float*)d_out;
  float* out_x   = out;                                 // [B,2,C]
  float* out_agg = out + B_ * CL * C_;                  // [B,N]
  float* out_idx = out_agg + B_ * N_;                   // [B,N]

  const size_t npad8 = (size_t)B_ * (N_ + 2) * C_ / 8;
  k_init   <<<(B_ * CL * C_ + 255) / 256, 256, 0, stream>>>(out_x, alw);
  k_xpad   <<<(int)((npad8 + 255) / 256), 256, 0, stream>>>(x, xpad);
  k_wprep  <<<(3 * C_ * C_ + 255) / 256, 256, 0, stream>>>(cw, wfrag);
  k_conv   <<<dim3(C_ / 32, N_ / 64, B_), 256, 0, stream>>>(x, xpad, wfrag, y);
  k_ln     <<<B_ * N_, 256, 0, stream>>>(y, g, be, sw, sb, xbf, tw, sq);
  k_density<<<dim3(N_ / 64, B_), 256, 0, stream>>>(xbf, sq, nz, dens, rmax);
  k_distmin<<<dim3(N_ / 64, B_), 256, 0, stream>>>(xbf, sq, dens, dmin);
  k_top2   <<<B_, 256, 0, stream>>>(rmax, dmin, dens, idn);
  k_assign <<<B_ * N_ / 8, 256, 0, stream>>>(y, sq, idn, tw, idc, alw);
  k_merge  <<<B_ * N_ / 256, 256, 0, stream>>>(y, tw, alw, idc, out_x, out_agg, out_idx);
}
